// ContactMapPredictor_68564857913754
// MI455X (gfx1250) — matrix operand order repaired and checked by reading, not compiled, not run on hardware
//
#include <hip/hip_runtime.h>
#include <cmath>

typedef _Float16 half_t;
typedef __attribute__((ext_vector_type(16))) _Float16 v16h;
typedef __attribute__((ext_vector_type(8)))  _Float16 v8h;
typedef __attribute__((ext_vector_type(8)))  float    v8f;
typedef unsigned int u32x4 __attribute__((ext_vector_type(4)));
typedef int          i32x8 __attribute__((ext_vector_type(8)));
typedef int          i32x4 __attribute__((ext_vector_type(4)));

#define LDS_STRIDE 40   // 32 halves (16 DWORDs) + 8 halves (4 DWORDs) pad = 80B/row

// ---- TDM: tensor_load_to_lds wrapper (CDNA5 ISA ch. 8 D# descriptor) ----

#if __has_include(<hip/amd_detail/amd_gfx1250_TDM.h>)
#define TDM_CALL(g0, g1, g2, g3, cpol)                                         \
    __builtin_amdgcn_tensor_load_to_lds((g0), (g1), (g2), (g3),                \
                                        (i32x8){0, 0, 0, 0, 0, 0, 0, 0}, (cpol))
#else
#define TDM_CALL(g0, g1, g2, g3, cpol)                                         \
    __builtin_amdgcn_tensor_load_to_lds((g0), (g1), (g2), (g3), (cpol))
#endif

__device__ __forceinline__ uint32_t lds_off_u32(const void* p) {
    return (uint32_t)(uintptr_t)(const __attribute__((address_space(3))) char*)p;
}

// 2D f16 tile load: tile1 rows x tile0 halves, row stride `stride0` halves,
// into LDS with +4-DWORD padding after every 16 DWORDs (i.e. LDS_STRIDE rows).
__device__ __forceinline__ void tdm_load_2d_f16(const half_t* gptr,
                                                uint32_t ldsOff,
                                                uint64_t stride0,
                                                uint32_t tile0, uint32_t tile1) {
    uint64_t gaddr = (uint64_t)(uintptr_t)gptr;
    const uint32_t dim0 = 1u << 20, dim1 = 1u << 20;  // no clipping; tiles in-bounds
    u32x4 g0;
    g0[0] = 1u;                                        // count=1, no gather
    g0[1] = ldsOff;                                    // lds_addr [63:32]
    g0[2] = (uint32_t)gaddr;                           // global_addr [95:64]
    g0[3] = ((uint32_t)(gaddr >> 32) & 0x1FFFFFFu) | (2u << 30);  // type=2
    i32x8 g1;
    g1[0] = (int)((1u << 16)      // data_size = 2B
                  | (1u << 20)    // pad_enable
                  | (3u << 22)    // pad_interval: 16 DWORDs
                  | (3u << 25));  // pad_amount:   4 DWORDs
    g1[1] = (int)((dim0 & 0xFFFFu) << 16);                              // dim0 lo16
    g1[2] = (int)(((dim0 >> 16) & 0xFFFFu) | ((dim1 & 0xFFFFu) << 16)); // dim0 hi/dim1 lo
    g1[3] = (int)(((dim1 >> 16) & 0xFFFFu) | ((tile0 & 0xFFFFu) << 16));// dim1 hi/tile0
    g1[4] = (int)(tile1 & 0xFFFFu);                                     // tile1, tile2=0
    g1[5] = (int)(uint32_t)(stride0 & 0xFFFFFFFFu);                     // stride0 lo32
    g1[6] = (int)(uint32_t)((stride0 >> 32) & 0xFFFFu);                 // stride0 hi16
    g1[7] = 0;
    i32x4 z4 = {0, 0, 0, 0};
    TDM_CALL(g0, g1, z4, z4, 0);
}

// ---- WMMA fragment loaders (CDNA5 ISA 7.12.2 layouts) -------------------

__device__ __forceinline__ v16h ldsFragA(const half_t* s, int rowBase) {
    int lane = threadIdx.x & 31;
    int m = lane & 15, hi = lane >> 4;
    const half_t* r = s + (rowBase + m) * LDS_STRIDE;
    v8h x0 = *(const v8h*)(r + hi * 8);
    v8h x1 = *(const v8h*)(r + 16 + hi * 8);
    v16h a;
#pragma unroll
    for (int i = 0; i < 8; ++i) { a[i] = x0[i]; a[i + 8] = x1[i]; }
    return a;
}

__device__ __forceinline__ v16h ldsFragB(const half_t* s, int colBase) {
    int lane = threadIdx.x & 31;
    int n = lane & 15, kOff = (lane >> 4) * 8;
    const half_t* r = s + (colBase + n) * LDS_STRIDE + kOff;
    v8h x0 = *(const v8h*)(r);
    v8h x1 = *(const v8h*)(r + 16);
    v16h b;
#pragma unroll
    for (int i = 0; i < 8; ++i) { b[i] = x0[i]; b[i + 8] = x1[i]; }
    return b;
}

__device__ __forceinline__ v8f wmma_f32_f16(v16h a, v16h b, v8f c) {
    return __builtin_amdgcn_wmma_f32_16x16x32_f16(false, a, false, b, (short)0, c,
                                                  false, false);
}

// ---- fp32 -> fp16 conversions ------------------------------------------

__global__ void cvt_f32_to_f16(const float* __restrict__ src,
                               half_t* __restrict__ dst, int n) {
    int i = blockIdx.x * blockDim.x + threadIdx.x;
    if (i < n) dst[i] = (half_t)src[i];
}

// w1 [512,128] fp32 -> w1T [128,512] f16 (so both halves become [N,K] rows)
__global__ void cvt_w1_transpose(const float* __restrict__ src,
                                 half_t* __restrict__ dst) {
    int i = blockIdx.x * blockDim.x + threadIdx.x;   // 512*128
    if (i < 512 * 128) {
        int k = i >> 7, n = i & 127;
        dst[(size_t)n * 512 + k] = (half_t)src[i];
    }
}

// ---- f16 WMMA GEMM with TDM double-buffered staging ---------------------
// C[M,N] = A[M,K] @ W^T + bias, W stored [N, ldw] row-major (torch Linear).
// Block = 128 threads (4 waves), tile 64x64, K-step 32.
__global__ __launch_bounds__(128) void gemm_wmma_f16(
    const half_t* __restrict__ A, const half_t* __restrict__ W,
    const float* __restrict__ bias, float* __restrict__ Cf,
    half_t* __restrict__ Ch, int M, int N, int K, int ldw) {
    __shared__ __align__(16) half_t As[2][64 * LDS_STRIDE];   // [row][k]
    __shared__ __align__(16) half_t Bs[2][64 * LDS_STRIDE];   // [col][k]
    const int t = threadIdx.x;
    const int n0 = blockIdx.x * 64, m0 = blockIdx.y * 64;
    const int wave = t >> 5, lane = t & 31;
    const int wRow = (wave >> 1) * 32, wCol = (wave & 1) * 32;
    const int nK = K >> 5;

    // prologue: wave 0 starts the DMA for K-tile 0
    if (t < 32) {
        tdm_load_2d_f16(A + (size_t)m0 * K, lds_off_u32(&As[0][0]),
                        (uint64_t)K, 32, 64);
        tdm_load_2d_f16(W + (size_t)n0 * ldw, lds_off_u32(&Bs[0][0]),
                        (uint64_t)ldw, 32, 64);
    }

    v8f acc[2][2] = {};
    for (int it = 0; it < nK; ++it) {
        const int cur = it & 1;
        if (t < 32) __builtin_amdgcn_s_wait_tensorcnt(0);
        __syncthreads();   // publish TDM data; also fences prior frag reads
        if (t < 32 && it + 1 < nK) {   // async-prefetch next K-tile
            int k1 = (it + 1) << 5;
            tdm_load_2d_f16(A + (size_t)m0 * K + k1, lds_off_u32(&As[cur ^ 1][0]),
                            (uint64_t)K, 32, 64);
            tdm_load_2d_f16(W + (size_t)n0 * ldw + k1, lds_off_u32(&Bs[cur ^ 1][0]),
                            (uint64_t)ldw, 32, 64);
        }
        v16h aF[2], bF[2];
        aF[0] = ldsFragA(As[cur], wRow);
        aF[1] = ldsFragA(As[cur], wRow + 16);
        bF[0] = ldsFragB(Bs[cur], wCol);
        bF[1] = ldsFragB(Bs[cur], wCol + 16);
#pragma unroll
        for (int mt = 0; mt < 2; ++mt)
#pragma unroll
            for (int nt = 0; nt < 2; ++nt)
                acc[mt][nt] = wmma_f32_f16(aF[mt], bF[nt], acc[mt][nt]);
    }

    // epilogue: D layout = VGPR r holds row (r + 8*hi), col = lane&15
    const int n = lane & 15, hi = lane >> 4;
#pragma unroll
    for (int mt = 0; mt < 2; ++mt)
#pragma unroll
        for (int nt = 0; nt < 2; ++nt) {
            int col = n0 + wCol + nt * 16 + n;
            float bv = bias ? bias[col] : 0.0f;
#pragma unroll
            for (int r = 0; r < 8; ++r) {
                int row = m0 + wRow + mt * 16 + hi * 8 + r;
                float v = acc[mt][nt][r] + bv;
                if (Ch) Ch[(size_t)row * N + col] = (half_t)v;
                else    Cf[(size_t)row * N + col] = v;
            }
        }
}

// ---- flash attention: one wave per (b, head, 16-query tile) -------------
__global__ __launch_bounds__(32) void attn_flash_wmma(
    const half_t* __restrict__ qkv, half_t* __restrict__ o) {
    __shared__ __align__(16) half_t kS[32 * LDS_STRIDE];   // [key][dim]
    __shared__ __align__(16) half_t vT[32 * LDS_STRIDE];   // [dim][key]
    __shared__ __align__(16) half_t pS[16 * LDS_STRIDE];   // [qrow][key]
    const int lane = threadIdx.x;
    const int qt = blockIdx.x, h = blockIdx.y, b = blockIdx.z;
    const int m = lane & 15, hi = lane >> 4;
    const size_t tokStride = 768;                    // 3H halves
    const size_t base = (size_t)b * 768 * tokStride;
    const float scale = 0.17677669529663687f;        // 1/sqrt(32)

    const half_t* qrow = qkv + base + (size_t)(qt * 16 + m) * tokStride + h * 32;
    v8h q0 = *(const v8h*)(qrow + hi * 8);
    v8h q1 = *(const v8h*)(qrow + 16 + hi * 8);
    v16h qF;
#pragma unroll
    for (int i = 0; i < 8; ++i) { qF[i] = q0[i]; qF[i + 8] = q1[i]; }

    v8f acc[2] = {};
    float mrow[8], lrow[8];
#pragma unroll
    for (int r = 0; r < 8; ++r) { mrow[r] = -1e30f; lrow[r] = 0.0f; }

    for (int kc = 0; kc < 768; kc += 32) {
        const half_t* kr = qkv + base + (size_t)(kc + lane) * tokStride + 256 + h * 32;
        const half_t* vr = qkv + base + (size_t)(kc + lane) * tokStride + 512 + h * 32;
        v8h kk0 = *(const v8h*)(kr), kk1 = *(const v8h*)(kr + 8);
        v8h kk2 = *(const v8h*)(kr + 16), kk3 = *(const v8h*)(kr + 24);
        *(v8h*)(kS + lane * LDS_STRIDE + 0)  = kk0;
        *(v8h*)(kS + lane * LDS_STRIDE + 8)  = kk1;
        *(v8h*)(kS + lane * LDS_STRIDE + 16) = kk2;
        *(v8h*)(kS + lane * LDS_STRIDE + 24) = kk3;
        v8h vv0 = *(const v8h*)(vr), vv1 = *(const v8h*)(vr + 8);
        v8h vv2 = *(const v8h*)(vr + 16), vv3 = *(const v8h*)(vr + 24);
#pragma unroll
        for (int d = 0; d < 8; ++d) {
            vT[(d)      * LDS_STRIDE + lane] = vv0[d];
            vT[(d + 8)  * LDS_STRIDE + lane] = vv1[d];
            vT[(d + 16) * LDS_STRIDE + lane] = vv2[d];
            vT[(d + 24) * LDS_STRIDE + lane] = vv3[d];
        }
        // single wave: DS ops are program-ordered, no barrier needed

        v8f sc[2];
        v8f zero = {};
        sc[0] = wmma_f32_f16(qF, ldsFragB(kS, 0), zero);
        sc[1] = wmma_f32_f16(qF, ldsFragB(kS, 16), zero);

        float cand[8];
#pragma unroll
        for (int r = 0; r < 8; ++r)
            cand[r] = fmaxf(sc[0][r], sc[1][r]) * scale;
#pragma unroll
        for (int msk = 1; msk < 16; msk <<= 1)
#pragma unroll
            for (int r = 0; r < 8; ++r)
                cand[r] = fmaxf(cand[r], __shfl_xor(cand[r], msk, 32));

        float psum[8];
        v8f p0, p1;
#pragma unroll
        for (int r = 0; r < 8; ++r) {
            float mn = fmaxf(mrow[r], cand[r]);
            float alpha = __expf(mrow[r] - mn);
            mrow[r] = mn;
            lrow[r] *= alpha;
            acc[0][r] *= alpha;
            acc[1][r] *= alpha;
            p0[r] = __expf(sc[0][r] * scale - mn);
            p1[r] = __expf(sc[1][r] * scale - mn);
            psum[r] = p0[r] + p1[r];
        }
#pragma unroll
        for (int msk = 1; msk < 16; msk <<= 1)
#pragma unroll
            for (int r = 0; r < 8; ++r)
                psum[r] += __shfl_xor(psum[r], msk, 32);
#pragma unroll
        for (int r = 0; r < 8; ++r) lrow[r] += psum[r];

#pragma unroll
        for (int r = 0; r < 8; ++r) {
            pS[(r + 8 * hi) * LDS_STRIDE + m]      = (half_t)p0[r];
            pS[(r + 8 * hi) * LDS_STRIDE + 16 + m] = (half_t)p1[r];
        }
        v16h pF = ldsFragA(pS, 0);
        acc[0] = wmma_f32_f16(pF, ldsFragB(vT, 0), acc[0]);
        acc[1] = wmma_f32_f16(pF, ldsFragB(vT, 16), acc[1]);
    }

#pragma unroll
    for (int nt = 0; nt < 2; ++nt)
#pragma unroll
        for (int r = 0; r < 8; ++r) {
            size_t tok = (size_t)b * 768 + qt * 16 + hi * 8 + r;
            o[tok * 256 + h * 32 + nt * 16 + m] = (half_t)(acc[nt][r] / lrow[r]);
        }
}

// ---- pairwise contact head: out[b,i,j] = relu(row_i + col'_j) . w2 + b2 --
__global__ __launch_bounds__(256) void pairwise_contact(
    const float* __restrict__ rowF, const float* __restrict__ colF,
    const float* __restrict__ w2, const float* __restrict__ b2,
    float* __restrict__ out) {
    __shared__ float rS[16][132];   // stride 132 floats -> conflict-free
    __shared__ float cS[16][132];
    __shared__ float wS[128];
    const int t = threadIdx.x;
    const int i0 = blockIdx.x * 16, j0 = blockIdx.y * 16, b = blockIdx.z;

    for (int c = t; c < 512; c += 256) {
        int r = c >> 5, off = (c & 31) * 4;
        *(float4*)&rS[r][off] =
            *(const float4*)(rowF + ((size_t)b * 768 + i0 + r) * 128 + off);
        *(float4*)&cS[r][off] =
            *(const float4*)(colF + ((size_t)b * 768 + j0 + r) * 128 + off);
    }
    if (t < 128) wS[t] = w2[t];
    __syncthreads();

    const int ti = t >> 4, tj = t & 15;
    float acc = 0.0f;
#pragma unroll 8
    for (int c = 0; c < 128; ++c)
        acc = fmaf(fmaxf(rS[ti][c] + cS[tj][c], 0.0f), wS[c], acc);
    out[((size_t)b * 768 + i0 + ti) * 768 + j0 + tj] = acc + b2[0];
}

// ------------------------------------------------------------------------

extern "C" void kernel_launch(void* const* d_in, const int* in_sizes, int n_in,
                              void* d_out, int out_size, void* d_ws, size_t ws_size,
                              hipStream_t stream) {
    (void)in_sizes; (void)n_in; (void)out_size; (void)ws_size;
    const int B = 2, L = 768, H = 256, M = B * L;       // M = 1536
    const float* features  = (const float*)d_in[0];
    const float* in_proj_w = (const float*)d_in[1];
    const float* in_proj_b = (const float*)d_in[2];
    const float* out_w     = (const float*)d_in[3];
    const float* out_b     = (const float*)d_in[4];
    const float* w1        = (const float*)d_in[5];
    const float* b1        = (const float*)d_in[6];
    const float* w2        = (const float*)d_in[7];
    const float* b2        = (const float*)d_in[8];
    float* out = (float*)d_out;

    char* ws = (char*)d_ws;
    size_t off = 0;
    auto alloc = [&](size_t bytes) -> void* {
        void* p = ws + off;
        off += (bytes + 255) & ~(size_t)255;
        return p;
    };
    half_t* featH = (half_t*)alloc((size_t)M * H * 2);          // [1536,256]
    half_t* wInH  = (half_t*)alloc((size_t)3 * H * H * 2);      // [768,256]
    half_t* outWH = (half_t*)alloc((size_t)H * H * 2);          // [256,256]
    half_t* w1T   = (half_t*)alloc((size_t)128 * 512 * 2);      // [128,512] = w1^T
    half_t* qkvH  = (half_t*)alloc((size_t)M * 3 * H * 2);      // [1536,768]
    half_t* oH    = (half_t*)alloc((size_t)M * H * 2);          // [1536,256]
    half_t* o2H   = (half_t*)alloc((size_t)M * H * 2);          // [1536,256]
    float*  rowF  = (float*)alloc((size_t)M * (H / 2) * 4);     // [1536,128]
    float*  colF  = (float*)alloc((size_t)M * (H / 2) * 4);     // [1536,128]

    auto cvt = [&](const float* s, half_t* d, int n) {
        cvt_f32_to_f16<<<(n + 255) / 256, 256, 0, stream>>>(s, d, n);
    };
    cvt(features,  featH, M * H);
    cvt(in_proj_w, wInH,  3 * H * H);
    cvt(out_w,     outWH, H * H);
    cvt_w1_transpose<<<(512 * 128 + 255) / 256, 256, 0, stream>>>(w1, w1T);

    // 1) QKV projection: [1536,768] = featH @ in_proj_w^T + in_proj_b
    gemm_wmma_f16<<<dim3(768 / 64, M / 64), 128, 0, stream>>>(
        featH, wInH, in_proj_b, nullptr, qkvH, M, 3 * H, H, H);

    // 2) flash attention per (b, head, 16-query tile)
    attn_flash_wmma<<<dim3(L / 16, 8, B), 32, 0, stream>>>(qkvH, oH);

    // 3) out projection: [1536,256] = oH @ out_w^T + out_b
    gemm_wmma_f16<<<dim3(H / 64, M / 64), 128, 0, stream>>>(
        oH, outWH, out_b, nullptr, o2H, M, H, H, H);

    // 4) row = o2 @ w1[:256]        (W = w1T rows, ldw=512)
    gemm_wmma_f16<<<dim3(128 / 64, M / 64), 128, 0, stream>>>(
        o2H, w1T, nullptr, rowF, nullptr, M, H / 2, H, 512);
    // 5) col = o2 @ w1[256:] + b1   (W = w1T rows offset by 256, ldw=512)
    gemm_wmma_f16<<<dim3(128 / 64, M / 64), 128, 0, stream>>>(
        o2H, w1T + 256, b1, colF, nullptr, M, H / 2, H, 512);

    // 6) pairwise contact map
    pairwise_contact<<<dim3(L / 16, L / 16, B), 256, 0, stream>>>(
        rowF, colF, w2, b2, out);
}
